// BasicTransDecoderBlock_54795192762526
// MI455X (gfx1250) — matrix operand order repaired and checked by reading, not compiled, not run on hardware
//
#include <hip/hip_runtime.h>
#include <hip/hip_bf16.h>

typedef __attribute__((ext_vector_type(16))) _Float16 v16h;
typedef __attribute__((ext_vector_type(8)))  float    v8f;
typedef int aint4 __attribute__((vector_size(16)));   // async-copy builtin operand type

#define IN_CH   256
#define OUT_CH  128
#define HEADS   4
#define DHD     32
#define N_TOK   4096   // 16^3
#define N_LO    512    // 8^3
#define TBL     29791  // (2*16-1)^3
#define EPS     1e-5f

#ifndef __has_builtin
#define __has_builtin(x) 0
#endif

#if __has_builtin(__builtin_amdgcn_global_load_async_to_lds_b128)
#define HAVE_ASYNC_COPY 1
#define ASYNC_CP_B128(gsrc, ldst) \
    __builtin_amdgcn_global_load_async_to_lds_b128((aint4*)(gsrc), (aint4*)(ldst), 0, 0)
#else
#define HAVE_ASYNC_COPY 0
#endif

__device__ inline void wait_async_all() {
#if __has_builtin(__builtin_amdgcn_s_wait_asynccnt)
    __builtin_amdgcn_s_wait_asynccnt(0);
#else
    asm volatile("s_wait_asynccnt 0x0" ::: "memory");
#endif
}

// ---------------------------------------------------------------------------
// Trilinear 8->16, align_corners=True
// ---------------------------------------------------------------------------
__device__ inline float interp8(const float* __restrict__ ch, int X, int Y, int Z) {
    const float s = 7.0f / 15.0f;
    float fx = X * s, fy = Y * s, fz = Z * s;
    int x0 = (int)fx, y0 = (int)fy, z0 = (int)fz;
    int x1 = min(x0 + 1, 7), y1 = min(y0 + 1, 7), z1 = min(z0 + 1, 7);
    float wx = fx - (float)x0, wy = fy - (float)y0, wz = fz - (float)z0;
    float c000 = ch[(x0 * 8 + y0) * 8 + z0];
    float c001 = ch[(x0 * 8 + y0) * 8 + z1];
    float c010 = ch[(x0 * 8 + y1) * 8 + z0];
    float c011 = ch[(x0 * 8 + y1) * 8 + z1];
    float c100 = ch[(x1 * 8 + y0) * 8 + z0];
    float c101 = ch[(x1 * 8 + y0) * 8 + z1];
    float c110 = ch[(x1 * 8 + y1) * 8 + z0];
    float c111 = ch[(x1 * 8 + y1) * 8 + z1];
    float c00 = c000 + (c001 - c000) * wz;
    float c01 = c010 + (c011 - c010) * wz;
    float c10 = c100 + (c101 - c100) * wz;
    float c11 = c110 + (c111 - c110) * wz;
    float c0 = c00 + (c01 - c00) * wy;
    float c1 = c10 + (c11 - c10) * wy;
    return c0 + (c1 - c0) * wx;
}

// ---------------------------------------------------------------------------
// 1x1x1 conv 256->128 @ 8^3 (+bias) -> y8
// ---------------------------------------------------------------------------
__global__ void conv_ch_kernel(const float* __restrict__ x1, const float* __restrict__ wch,
                               const float* __restrict__ bch, float* __restrict__ y8) {
    int gid = blockIdx.x * blockDim.x + threadIdx.x;   // 128*512
    int oc = gid >> 9, p = gid & 511;
    float acc = bch[oc];
    const float* wr = wch + (size_t)oc * IN_CH;
    for (int ic = 0; ic < IN_CH; ++ic) acc += wr[ic] * x1[(size_t)ic * N_LO + p];
    y8[gid] = acc;
}

// residue = interp(y8) to 16^3
__global__ void interp_res_kernel(const float* __restrict__ y8, float* __restrict__ residue) {
    int gid = blockIdx.x * blockDim.x + threadIdx.x;   // 128*4096
    int c = gid >> 12, p = gid & 4095;
    int X = p >> 8, Y = (p >> 4) & 15, Z = p & 15;
    residue[gid] = interp8(y8 + (size_t)c * N_LO, X, Y, Z);
}

// ---------------------------------------------------------------------------
// Training-mode BN stats: one block per channel
// ---------------------------------------------------------------------------
__global__ void bn_stats_kernel(const float* __restrict__ x, float* __restrict__ mean,
                                float* __restrict__ istd, int S) {
    int c = blockIdx.x;
    const float* xc = x + (size_t)c * S;
    float s = 0.f, ss = 0.f;
    for (int i = threadIdx.x; i < S; i += 256) { float v = xc[i]; s += v; ss += v * v; }
    __shared__ float rs[256], rq[256];
    rs[threadIdx.x] = s; rq[threadIdx.x] = ss;
    __syncthreads();
    for (int o = 128; o > 0; o >>= 1) {
        if (threadIdx.x < o) { rs[threadIdx.x] += rs[threadIdx.x + o]; rq[threadIdx.x] += rq[threadIdx.x + o]; }
        __syncthreads();
    }
    if (threadIdx.x == 0) {
        float m = rs[0] / (float)S;
        float v = rq[0] / (float)S - m * m;
        mean[c] = m;
        istd[c] = rsqrtf(v + EPS);
    }
}

// ---------------------------------------------------------------------------
// Depthwise 3x3x3 conv with BN applied to the input on the fly (pad=1)
// ---------------------------------------------------------------------------
__global__ void dw3_bn_kernel(const float* __restrict__ x, const float* __restrict__ wdw,
                              const float* __restrict__ mean, const float* __restrict__ istd,
                              const float* __restrict__ gamma, const float* __restrict__ beta,
                              float* __restrict__ out, int Dm) {
    int S = Dm * Dm * Dm;
    int gid = blockIdx.x * blockDim.x + threadIdx.x;
    int c = gid / S, p = gid % S;
    int Z = p % Dm, Y = (p / Dm) % Dm, X = p / (Dm * Dm);
    float m = mean[c], is = istd[c], g = gamma[c], b = beta[c];
    const float* xc = x + (size_t)c * S;
    const float* wc = wdw + (size_t)c * 27;
    float acc = 0.f;
    for (int dx = -1; dx <= 1; ++dx) {
        int xx = X + dx; if ((unsigned)xx >= (unsigned)Dm) continue;
        for (int dy = -1; dy <= 1; ++dy) {
            int yy = Y + dy; if ((unsigned)yy >= (unsigned)Dm) continue;
            for (int dz = -1; dz <= 1; ++dz) {
                int zz = Z + dz; if ((unsigned)zz >= (unsigned)Dm) continue;
                float v = (xc[(xx * Dm + yy) * Dm + zz] - m) * is * g + b;
                acc += wc[((dx + 1) * 3 + (dy + 1)) * 3 + (dz + 1)] * v;
            }
        }
    }
    out[gid] = acc;
}

// ---------------------------------------------------------------------------
// Pointwise convs
// ---------------------------------------------------------------------------
__global__ void pw_kv_kernel(const float* __restrict__ dwl, const float* __restrict__ wpw,
                             float* __restrict__ kv8) {
    int gid = blockIdx.x * blockDim.x + threadIdx.x;   // 256*512
    int oc = gid >> 9, p = gid & 511;
    const float* wr = wpw + (size_t)oc * IN_CH;
    float acc = 0.f;
    for (int ic = 0; ic < IN_CH; ++ic) acc += wr[ic] * dwl[(size_t)ic * N_LO + p];
    kv8[gid] = acc;
}

// interp k,v 8^3 -> 16^3 and pack f16: k -> [head][tok][d], v -> transposed [head][d][tok]
__global__ void kv_interp_pack_kernel(const float* __restrict__ kv8,
                                      _Float16* __restrict__ kf, _Float16* __restrict__ vt) {
    int gid = blockIdx.x * blockDim.x + threadIdx.x;   // 2*128*4096 = 2^20
    int which = gid >> 19;                             // 0=k, 1=v
    int rem = gid & ((1 << 19) - 1);
    int c = rem >> 12, p = rem & 4095;
    int X = p >> 8, Y = (p >> 4) & 15, Z = p & 15;
    float val = interp8(kv8 + (size_t)(which * OUT_CH + c) * N_LO, X, Y, Z);
    int head = c & 3, dh = c >> 2;                     // channel layout (dim_head, heads)
    if (which == 0) kf[((size_t)(head * N_TOK + p)) * DHD + dh] = (_Float16)val;
    else            vt[((size_t)(head * DHD + dh)) * N_TOK + p] = (_Float16)val;
}

// pointwise q (128->128) + pack f16 [head][tok][d]
__global__ void pw_q_pack_kernel(const float* __restrict__ dwh, const float* __restrict__ wpw,
                                 _Float16* __restrict__ qf) {
    int gid = blockIdx.x * blockDim.x + threadIdx.x;   // 128*4096
    int oc = gid >> 12, p = gid & 4095;
    const float* wr = wpw + (size_t)oc * OUT_CH;
    float acc = 0.f;
    for (int ic = 0; ic < OUT_CH; ++ic) acc += wr[ic] * dwh[(size_t)ic * N_TOK + p];
    int head = oc & 3, dh = oc >> 2;
    qf[((size_t)(head * N_TOK + p)) * DHD + dh] = (_Float16)acc;
}

// ---------------------------------------------------------------------------
// Flash attention with WMMA f32_16x16x32_f16.
// 4 waves/block, all sharing one head; each wave owns a 16-query tile.
// K/V chunks (32 keys) are double-buffered in LDS via async global->LDS copies
// (all 4 waves consume the same chunk, so it's fetched once per block).
// Relative-position bias reduces to 2 integer bases per chunk (no div/mod).
// ---------------------------------------------------------------------------
__global__ __launch_bounds__(128) void attn_kernel(
        const _Float16* __restrict__ qf, const _Float16* __restrict__ kf,
        const _Float16* __restrict__ vt, const float* __restrict__ rel_table,
        float* __restrict__ o) {
    __shared__ __align__(64) float    sS[4][16 * 32];
    __shared__ __align__(64) _Float16 sP[4][32 * 16];
    __shared__ __align__(64) float    sR[4][16];
#if HAVE_ASYNC_COPY
    __shared__ __align__(128) _Float16 kTile[2][32][DHD];   // [key_local][dim]
    __shared__ __align__(128) _Float16 vTile[2][32][32];    // [dim][key_local]
#endif

    const int tid  = threadIdx.x;
    const int wv   = tid >> 5;
    const int lane = tid & 31;
    const int hi   = lane >> 4;        // K-block half within wave
    const int mr   = lane & 15;        // row (A/C) or column (B)

    int t     = blockIdx.x * 4 + wv;   // 0..1023; head uniform across block
    int head  = t >> 8;
    int qtile = t & 255;
    int q0    = qtile << 4;

    const _Float16* kb_base = kf + (size_t)head * N_TOK * DHD;
    const _Float16* vb_base = vt + (size_t)head * DHD * N_TOK;

    // ---- Q fragment, A layout 16x32 f16 ----
    v16h qa;
    {
        const _Float16* qrow = qf + ((size_t)(head * N_TOK + q0 + mr)) * DHD;
#pragma unroll
        for (int j = 0; j < 16; ++j) {
            int kd = hi * 16 + j;
            qa[j] = qrow[kd];
        }
    }

    v8f o0 = {0.f, 0.f, 0.f, 0.f, 0.f, 0.f, 0.f, 0.f};
    v8f o1 = {0.f, 0.f, 0.f, 0.f, 0.f, 0.f, 0.f, 0.f};
    const v8f zc = {0.f, 0.f, 0.f, 0.f, 0.f, 0.f, 0.f, 0.f};

    // softmax running state: lanes 0..15 own row == lane
    float rm = -3.0e38f, rl = 0.f;
    int qrow_g = q0 + mr;
    int qh = qrow_g >> 8, qw = (qrow_g >> 4) & 15, qd = qrow_g & 15;
    const float scale = 0.17677669529663687f;   // 32^-0.5

#if HAVE_ASYNC_COPY
    // cooperative async copy: 128 threads, one b128 each into K tile and V tile
    const int crow  = tid >> 2;        // 0..31
    const int cpart = (tid & 3) * 8;   // 0,8,16,24 halfs (16B each)
    {   // prologue: chunk 0 -> buffer 0
        ASYNC_CP_B128(kb_base + (size_t)crow * DHD + cpart, &kTile[0][crow][cpart]);
        ASYNC_CP_B128(vb_base + (size_t)crow * N_TOK + cpart, &vTile[0][crow][cpart]);
    }
#endif

    int it = 0;
    for (int kb = 0; kb < N_TOK; kb += 32, ++it) {
#if HAVE_ASYNC_COPY
        const int buf = it & 1;
        wait_async_all();              // this wave's share of current chunk is in LDS
        __syncthreads();               // everyone's share is; previous buffer free
        if (kb + 32 < N_TOK) {
            ASYNC_CP_B128(kb_base + (size_t)(kb + 32 + crow) * DHD + cpart,
                          &kTile[buf ^ 1][crow][cpart]);
            ASYNC_CP_B128(vb_base + (size_t)crow * N_TOK + (kb + 32) + cpart,
                          &vTile[buf ^ 1][crow][cpart]);
        }
        v16h kf0 = *(const v16h*)(&kTile[buf][mr][hi * 16]);
        v16h kf1 = *(const v16h*)(&kTile[buf][16 + mr][hi * 16]);
#else
        if (kb + 32 < N_TOK) {   // stream next chunk -> global_prefetch_b8
            __builtin_prefetch(kb_base + (size_t)(kb + 32) * DHD, 0, 1);
            __builtin_prefetch(vb_base + (kb + 32), 0, 1);
        }
        v16h kf0 = *(const v16h*)(kb_base + (size_t)(kb + mr) * DHD + hi * 16);
        v16h kf1 = *(const v16h*)(kb_base + (size_t)(kb + 16 + mr) * DHD + hi * 16);
#endif

        v8f s0 = __builtin_amdgcn_wmma_f32_16x16x32_f16(false, qa, false, kf0, (short)0, zc, false, false);
        v8f s1 = __builtin_amdgcn_wmma_f32_16x16x32_f16(false, qa, false, kf1, (short)0, zc, false, false);

        // dump scores (C layout) to LDS: lane holds col n=mr, rows r+8*hi
#pragma unroll
        for (int r = 0; r < 8; ++r) {
            sS[wv][(r + 8 * hi) * 32 + mr]      = s0[r];
            sS[wv][(r + 8 * hi) * 32 + 16 + mr] = s1[r];
        }
        asm volatile("s_wait_dscnt 0x0" ::: "memory");

        // per-row online softmax (lanes 0..15, one row each)
        if (lane < 16) {
            int row = lane;
            // the 32 keys of this chunk span exactly two (kh,kw) groups
            int base0 = (qh - (kb >> 8) + 15) * 31 + (qw - ((kb >> 4) & 15) + 15) + qd;
            int kb1i  = kb + 16;
            int base1 = (qh - (kb1i >> 8) + 15) * 31 + (qw - ((kb1i >> 4) & 15) + 15) + qd;
            float u[32];
            float cmax = -3.0e38f;
#pragma unroll
            for (int kk = 0; kk < 32; ++kk) {
                int idx = ((kk < 16) ? base0 : base1) - (kk & 15);
                idx += (idx >> 31) & TBL;              // torch-faithful negative wrap
                float uu = sS[wv][row * 32 + kk] + rel_table[idx * HEADS + head];
                u[kk] = uu;
                cmax = fmaxf(cmax, uu);
            }
            float nm   = fmaxf(rm, cmax);
            float resc = __expf((rm - nm) * scale);
            rl *= resc;
#pragma unroll
            for (int kk = 0; kk < 32; ++kk) {
                float p = __expf((u[kk] - nm) * scale);
                rl += p;
                int hi2, j2;                           // C-layout -> A-layout transpose
                if (kk < 16) { hi2 = 0; j2 = kk; }
                else         { hi2 = 1; j2 = kk - 16; }
                sP[wv][(row + 16 * hi2) * 16 + j2] = (_Float16)p;
            }
            sR[wv][row] = resc;
            rm = nm;
        }
        asm volatile("s_wait_dscnt 0x0" ::: "memory");

        // rescale running O by per-row factor
#pragma unroll
        for (int r = 0; r < 8; ++r) {
            float f = sR[wv][r + 8 * hi];
            o0[r] *= f; o1[r] *= f;
        }
        // P fragment (A layout) back from LDS, V fragments (B layout)
        v16h pa = *(const v16h*)(&sP[wv][lane * 16]);
#if HAVE_ASYNC_COPY
        v16h vf0 = *(const v16h*)(&vTile[buf][mr][hi * 16]);
        v16h vf1 = *(const v16h*)(&vTile[buf][16 + mr][hi * 16]);
#else
        v16h vf0 = *(const v16h*)(vb_base + (size_t)(mr) * N_TOK + kb + hi * 16);
        v16h vf1 = *(const v16h*)(vb_base + (size_t)(16 + mr) * N_TOK + kb + hi * 16);
#endif
        o0 = __builtin_amdgcn_wmma_f32_16x16x32_f16(false, pa, false, vf0, (short)0, o0, false, false);
        o1 = __builtin_amdgcn_wmma_f32_16x16x32_f16(false, pa, false, vf1, (short)0, o1, false, false);
    }

    // finalize: divide by row sums, write [head][tok][d] f32
    if (lane < 16) sR[wv][lane] = rl;
    asm volatile("s_wait_dscnt 0x0" ::: "memory");
#pragma unroll
    for (int r = 0; r < 8; ++r) {
        int row   = r + 8 * hi;
        float inv = 1.f / sR[wv][row];
        size_t base = ((size_t)(head * N_TOK + q0 + row)) * DHD;
        o[base + mr]      = o0[r] * inv;
        o[base + 16 + mr] = o1[r] * inv;
    }
}

// ---------------------------------------------------------------------------
// Output path
// ---------------------------------------------------------------------------
// depthwise 3x3x3 over the attention output (remapped from [head][tok][d])
__global__ void out_dw_kernel(const float* __restrict__ o, const float* __restrict__ wdw,
                              float* __restrict__ out) {
    int gid = blockIdx.x * blockDim.x + threadIdx.x;   // 128*4096
    int c = gid >> 12, p = gid & 4095;
    int head = c & 3, dh = c >> 2;
    int X = p >> 8, Y = (p >> 4) & 15, Z = p & 15;
    const float* wc = wdw + (size_t)c * 27;
    float acc = 0.f;
    for (int dx = -1; dx <= 1; ++dx) {
        int xx = X + dx; if ((unsigned)xx >= 16u) continue;
        for (int dy = -1; dy <= 1; ++dy) {
            int yy = Y + dy; if ((unsigned)yy >= 16u) continue;
            for (int dz = -1; dz <= 1; ++dz) {
                int zz = Z + dz; if ((unsigned)zz >= 16u) continue;
                int pp = (xx << 8) | (yy << 4) | zz;
                acc += wc[((dx + 1) * 3 + (dy + 1)) * 3 + (dz + 1)]
                     * o[((size_t)(head * N_TOK + pp)) * DHD + dh];
            }
        }
    }
    out[gid] = acc;
}

// pointwise 128->128 + residue  ->  res2
__global__ void pw_out_res_kernel(const float* __restrict__ odw, const float* __restrict__ wpw,
                                  const float* __restrict__ residue, float* __restrict__ res2) {
    int gid = blockIdx.x * blockDim.x + threadIdx.x;   // 128*4096
    int oc = gid >> 12, p = gid & 4095;
    const float* wr = wpw + (size_t)oc * OUT_CH;
    float acc = residue[gid];
    for (int ic = 0; ic < OUT_CH; ++ic) acc += wr[ic] * odw[(size_t)ic * N_TOK + p];
    res2[gid] = acc;
}

// out = w_mlp @ relu(bn(res2)) + res2
__global__ void final_kernel(const float* __restrict__ res2, const float* __restrict__ wmlp,
                             const float* __restrict__ mean2, const float* __restrict__ istd2,
                             const float* __restrict__ g2, const float* __restrict__ b2,
                             float* __restrict__ out) {
    int gid = blockIdx.x * blockDim.x + threadIdx.x;   // 128*4096
    int oc = gid >> 12, p = gid & 4095;
    const float* wr = wmlp + (size_t)oc * OUT_CH;
    float acc = 0.f;
    for (int ic = 0; ic < OUT_CH; ++ic) {
        float tv = res2[(size_t)ic * N_TOK + p];
        tv = (tv - mean2[ic]) * istd2[ic] * g2[ic] + b2[ic];
        tv = fmaxf(tv, 0.f);
        acc += wr[ic] * tv;
    }
    out[gid] = acc + res2[gid];
}

// ---------------------------------------------------------------------------
extern "C" void kernel_launch(void* const* d_in, const int* in_sizes, int n_in,
                              void* d_out, int out_size, void* d_ws, size_t ws_size,
                              hipStream_t stream) {
    (void)in_sizes; (void)n_in; (void)out_size; (void)ws_size;
    const float* x1       = (const float*)d_in[0];
    const float* x2       = (const float*)d_in[1];
    const float* w_ch     = (const float*)d_in[2];
    const float* b_ch     = (const float*)d_in[3];
    const float* gamma_l  = (const float*)d_in[4];
    const float* beta_l   = (const float*)d_in[5];
    const float* gamma_h  = (const float*)d_in[6];
    const float* beta_h   = (const float*)d_in[7];
    const float* gamma2   = (const float*)d_in[8];
    const float* beta2    = (const float*)d_in[9];
    const float* kv_dw    = (const float*)d_in[10];
    const float* kv_pw    = (const float*)d_in[11];
    const float* q_dw     = (const float*)d_in[12];
    const float* q_pw     = (const float*)d_in[13];
    const float* w_odw    = (const float*)d_in[14];
    const float* w_opw    = (const float*)d_in[15];
    const float* w_mlp    = (const float*)d_in[16];
    const float* rel_tab  = (const float*)d_in[17];
    float* out = (float*)d_out;

    char* wp = (char*)d_ws;
    auto alloc = [&](size_t n) { char* p = wp; wp += (n + 255) & ~(size_t)255; return p; };
    float*    y8      = (float*)alloc((size_t)OUT_CH * N_LO * 4);
    float*    residue = (float*)alloc((size_t)OUT_CH * N_TOK * 4);
    float*    mean_l  = (float*)alloc(IN_CH * 4);
    float*    istd_l  = (float*)alloc(IN_CH * 4);
    float*    mean_h  = (float*)alloc(OUT_CH * 4);
    float*    istd_h  = (float*)alloc(OUT_CH * 4);
    float*    mean2   = (float*)alloc(OUT_CH * 4);
    float*    istd2   = (float*)alloc(OUT_CH * 4);
    float*    dwl     = (float*)alloc((size_t)IN_CH * N_LO * 4);
    float*    dwh     = (float*)alloc((size_t)OUT_CH * N_TOK * 4);
    float*    kv8     = (float*)alloc((size_t)2 * OUT_CH * N_LO * 4);
    _Float16* qf      = (_Float16*)alloc((size_t)HEADS * N_TOK * DHD * 2);
    _Float16* kf      = (_Float16*)alloc((size_t)HEADS * N_TOK * DHD * 2);
    _Float16* vt      = (_Float16*)alloc((size_t)HEADS * DHD * N_TOK * 2);
    float*    o_ws    = (float*)alloc((size_t)HEADS * N_TOK * DHD * 4);
    float*    odw     = (float*)alloc((size_t)OUT_CH * N_TOK * 4);
    float*    res2    = (float*)alloc((size_t)OUT_CH * N_TOK * 4);

    dim3 b256(256);
    conv_ch_kernel       <<<(OUT_CH * N_LO) / 256,      b256, 0, stream>>>(x1, w_ch, b_ch, y8);
    interp_res_kernel    <<<(OUT_CH * N_TOK) / 256,     b256, 0, stream>>>(y8, residue);
    bn_stats_kernel      <<<IN_CH,                      b256, 0, stream>>>(x1, mean_l, istd_l, N_LO);
    bn_stats_kernel      <<<OUT_CH,                     b256, 0, stream>>>(x2, mean_h, istd_h, N_TOK);
    dw3_bn_kernel        <<<(IN_CH * N_LO) / 256,       b256, 0, stream>>>(x1, kv_dw, mean_l, istd_l, gamma_l, beta_l, dwl, 8);
    dw3_bn_kernel        <<<(OUT_CH * N_TOK) / 256,     b256, 0, stream>>>(x2, q_dw, mean_h, istd_h, gamma_h, beta_h, dwh, 16);
    pw_kv_kernel         <<<(2 * OUT_CH * N_LO) / 256,  b256, 0, stream>>>(dwl, kv_pw, kv8);
    kv_interp_pack_kernel<<<(2 * OUT_CH * N_TOK) / 256, b256, 0, stream>>>(kv8, kf, vt);
    pw_q_pack_kernel     <<<(OUT_CH * N_TOK) / 256,     b256, 0, stream>>>(dwh, q_pw, qf);
    attn_kernel          <<<256, dim3(128),             0, stream>>>(qf, kf, vt, rel_tab, o_ws);
    out_dw_kernel        <<<(OUT_CH * N_TOK) / 256,     b256, 0, stream>>>(o_ws, w_odw, odw);
    pw_out_res_kernel    <<<(OUT_CH * N_TOK) / 256,     b256, 0, stream>>>(odw, w_opw, residue, res2);
    bn_stats_kernel      <<<OUT_CH,                     b256, 0, stream>>>(res2, mean2, istd2, N_TOK);
    final_kernel         <<<(OUT_CH * N_TOK) / 256,     b256, 0, stream>>>(res2, w_mlp, mean2, istd2, gamma2, beta2, out);
}
